// NeuralField_68590627717362
// MI455X (gfx1250) — compile-verified, matrix operand order checked by reading, not run on hardware
//
#include <hip/hip_runtime.h>
#include <cmath>

typedef __attribute__((ext_vector_type(16))) _Float16 v16h;
typedef __attribute__((ext_vector_type(8)))  _Float16 v8h;
typedef __attribute__((ext_vector_type(8)))  float    v8f;

#define N_LEVELS 10
#define HASH_SIZE (1u << 20)
#define TILE_M   64
#define THREADS  128

// W1 swizzled: 3 kc * 24 nt * (32 lanes * 16 halves) = 36864 halves
#define W1S_HALVES (3 * 24 * 512)
// W2 swizzled: 12 kc * 48 nt * 512 = 294912 halves
#define W2S_HALVES (12 * 48 * 512)

struct LevelParams {
    float    scale[N_LEVELS];
    unsigned res[N_LEVELS];
    unsigned dense[N_LEVELS];
};

// ---------------------------------------------------------------------------
// Weight prep: f32 -> f16, pre-swizzled into WMMA B-fragment order.
// B fragment layout (16-bit B, 32x16 K x N): lane l holds column n = l&15;
// the 16 packed halves map to K via: half = l>>4, v = j>>1,
//   klo = (v<4) ? 2v : 16+2(v-4),  k = klo + 8*half + (j&1).
// ---------------------------------------------------------------------------
__global__ void prep_weights(const float* __restrict__ w1,
                             const float* __restrict__ w2,
                             _Float16* __restrict__ W1s,
                             _Float16* __restrict__ W2s) {
    int tid = blockIdx.x * 256 + threadIdx.x;
    if (tid < W1S_HALVES) {
        int j = tid & 15, lane = (tid >> 4) & 31, frag = tid >> 9;
        int kc = frag / 24, nt = frag - kc * 24;
        int hlf = lane >> 4, v = j >> 1;
        int klo = (v < 4) ? 2 * v : 16 + 2 * (v - 4);
        int k  = kc * 32 + klo + 8 * hlf + (j & 1);
        int nn = nt * 16 + (lane & 15);
        W1s[tid] = (_Float16)((k < 80) ? w1[k * 384 + nn] : 0.0f);
    } else if (tid < W1S_HALVES + W2S_HALVES) {
        int t2 = tid - W1S_HALVES;
        int j = t2 & 15, lane = (t2 >> 4) & 31, frag = t2 >> 9;
        int kc = frag / 48, nt = frag - kc * 48;
        int hlf = lane >> 4, v = j >> 1;
        int klo = (v < 4) ? 2 * v : 16 + 2 * (v - 4);
        int k  = kc * 32 + klo + 8 * hlf + (j & 1);
        int nn = nt * 16 + (lane & 15);
        W2s[t2] = (_Float16)w2[k * 768 + nn];
    }
}

// A fragment (16-bit A, 16x32 M x K): lane l holds row m = l&15; the 16 halves
// are two contiguous K-runs of 8: k in [k0, k0+8) and [k0+16, k0+24), with
// k0 = kc*32 + 8*(l>>4).  Two 16-byte LDS loads per fragment.
__device__ __forceinline__ v16h load_a_frag(const _Float16* s, int row, int ldk, int k0) {
    const _Float16* p = s + row * ldk + k0;
    v8h lo = *(const v8h*)(p);
    v8h hi = *(const v8h*)(p + 16);
    v16h a;
#pragma unroll
    for (int i = 0; i < 8; ++i) { a[i] = lo[i]; a[i + 8] = hi[i]; }
    return a;
}

__global__ __launch_bounds__(THREADS)
void nf_fused(const float* __restrict__ coords,
              const float* __restrict__ table,
              const float* __restrict__ b1,
              const float* __restrict__ b2,
              const _Float16* __restrict__ W1s,
              const _Float16* __restrict__ W2s,
              float* __restrict__ out,
              LevelParams lp) {
    __shared__ _Float16 sEnc[TILE_M * 96];    // 12 KB, K padded 80 -> 96
    __shared__ _Float16 sH[TILE_M * 384];     // 48 KB

    const int t = threadIdx.x;

    // ---------------- Phase 1: hash-grid encoding -> sEnc (f16) -------------
    {
        const int p    = t >> 1;            // 2 threads per point
        const int lset = (t & 1) * 5;       // levels 0-4 or 5-9
        const int pt   = blockIdx.x * TILE_M + p;
        const float cx = coords[2 * pt + 0];
        const float cy = coords[2 * pt + 1];
#pragma unroll
        for (int li = 0; li < 5; ++li) {
            const int l = lset + li;
            const float    scale = lp.scale[l];
            const unsigned res   = lp.res[l];
            const unsigned dense = lp.dense[l];
            const float px = cx * scale + 0.5f;
            const float py = cy * scale + 0.5f;
            const float fx = floorf(px), fy = floorf(py);
            const float tx = px - fx, ty = py - fy;
            const unsigned x0 = (unsigned)fx, y0 = (unsigned)fy;
            const float* tab = table + (size_t)l * (size_t)HASH_SIZE * 8u;
            float acc[8];
#pragma unroll
            for (int f = 0; f < 8; ++f) acc[f] = 0.0f;
#pragma unroll
            for (int c = 0; c < 4; ++c) {
                const unsigned xx = x0 + (c & 1);
                const unsigned yy = y0 + (c >> 1);
                const unsigned idx = dense
                    ? (xx + yy * res)
                    : (((xx * 1u) ^ (yy * 2654435761u)) & (HASH_SIZE - 1u));
                const float wx = (c & 1) ? tx : 1.0f - tx;
                const float wy = (c & 2) ? ty : 1.0f - ty;
                const float w  = wx * wy;
                const float4 lo = *(const float4*)(tab + (size_t)idx * 8u);
                const float4 hi = *(const float4*)(tab + (size_t)idx * 8u + 4u);
                acc[0] = fmaf(w, lo.x, acc[0]); acc[1] = fmaf(w, lo.y, acc[1]);
                acc[2] = fmaf(w, lo.z, acc[2]); acc[3] = fmaf(w, lo.w, acc[3]);
                acc[4] = fmaf(w, hi.x, acc[4]); acc[5] = fmaf(w, hi.y, acc[5]);
                acc[6] = fmaf(w, hi.z, acc[6]); acc[7] = fmaf(w, hi.w, acc[7]);
            }
#pragma unroll
            for (int f = 0; f < 8; ++f)
                sEnc[p * 96 + l * 8 + f] = (_Float16)acc[f];
        }
        if (t & 1) {                        // zero-pad K 80..95
#pragma unroll
            for (int f = 80; f < 96; ++f) sEnc[(t >> 1) * 96 + f] = (_Float16)0.0f;
        }
    }
    __syncthreads();

    const int lane = t & 31;
    const int wave = t >> 5;                // m-tile owned by this wave (0..3)
    const int hlf  = lane >> 4;
    const int ncol = lane & 15;
    const int mrow = wave * 16 + ncol;      // A-fragment source row

    // ------- Phase 2: GEMM1 [64x96]x[96x384] + bias + ReLU -> sH (f16) ------
    {
        v16h a0 = load_a_frag(sEnc, mrow, 96, 0 * 32 + 8 * hlf);
        v16h a1 = load_a_frag(sEnc, mrow, 96, 1 * 32 + 8 * hlf);
        v16h a2 = load_a_frag(sEnc, mrow, 96, 2 * 32 + 8 * hlf);
        for (int nt = 0; nt < 24; ++nt) {
            v8f acc = {};
            v16h bw0 = *(const v16h*)(W1s + ((size_t)(0 * 24 + nt) * 32 + lane) * 16);
            acc = __builtin_amdgcn_wmma_f32_16x16x32_f16(false, a0, false, bw0,
                                                         (short)0, acc, false, false);
            v16h bw1 = *(const v16h*)(W1s + ((size_t)(1 * 24 + nt) * 32 + lane) * 16);
            acc = __builtin_amdgcn_wmma_f32_16x16x32_f16(false, a1, false, bw1,
                                                         (short)0, acc, false, false);
            v16h bw2 = *(const v16h*)(W1s + ((size_t)(2 * 24 + nt) * 32 + lane) * 16);
            acc = __builtin_amdgcn_wmma_f32_16x16x32_f16(false, a2, false, bw2,
                                                         (short)0, acc, false, false);
            const float bias = b1[nt * 16 + ncol];
#pragma unroll
            for (int r = 0; r < 8; ++r) {
                float v = acc[r] + bias;
                v = v > 0.0f ? v : 0.0f;    // ReLU
                sH[(wave * 16 + 8 * hlf + r) * 384 + nt * 16 + ncol] = (_Float16)v;
            }
        }
    }
    __syncthreads();

    // --------- Phase 3: GEMM2 [64x384]x[384x768] + bias -> out (f32) --------
    {
        v16h afr[12];
#pragma unroll
        for (int kc = 0; kc < 12; ++kc)
            afr[kc] = load_a_frag(sH, mrow, 384, kc * 32 + 8 * hlf);
        const int gm0 = blockIdx.x * TILE_M + wave * 16 + 8 * hlf;
        for (int nt = 0; nt < 48; ++nt) {
            v8f acc = {};
#pragma unroll
            for (int kc = 0; kc < 12; ++kc) {
                v16h bw = *(const v16h*)(W2s + ((size_t)(kc * 48 + nt) * 32 + lane) * 16);
                acc = __builtin_amdgcn_wmma_f32_16x16x32_f16(false, afr[kc], false, bw,
                                                             (short)0, acc, false, false);
            }
            const int col = nt * 16 + ncol;
            const float bias = b2[col];
#pragma unroll
            for (int r = 0; r < 8; ++r)
                out[(size_t)(gm0 + r) * 768 + col] = acc[r] + bias;
        }
    }
}

extern "C" void kernel_launch(void* const* d_in, const int* in_sizes, int n_in,
                              void* d_out, int out_size, void* d_ws, size_t ws_size,
                              hipStream_t stream) {
    const float* coords = (const float*)d_in[0];
    const float* table  = (const float*)d_in[1];
    const float* w1     = (const float*)d_in[2];
    const float* b1     = (const float*)d_in[3];
    const float* w2     = (const float*)d_in[4];
    const float* b2     = (const float*)d_in[5];
    float* out = (float*)d_out;
    const int n = in_sizes[0] / 2;          // number of points (131072)

    _Float16* W1s = (_Float16*)d_ws;
    _Float16* W2s = (_Float16*)d_ws + W1S_HALVES;

    // Per-level scale / resolution, computed in double exactly like the
    // reference (numpy): growth g = exp((ln 1024 - ln 16)/9).
    LevelParams lp;
    const double g = exp((log(1024.0) - log(16.0)) / 9.0);
    for (int l = 0; l < N_LEVELS; ++l) {
        double s = 16.0 * pow(g, (double)l) - 1.0;
        int res  = (int)ceil(s) + 1;
        lp.scale[l] = (float)s;
        lp.res[l]   = (unsigned)res;
        lp.dense[l] = ((unsigned long long)res * (unsigned long long)res
                       <= (unsigned long long)HASH_SIZE) ? 1u : 0u;
    }

    const int prep_threads = W1S_HALVES + W2S_HALVES;         // 331776
    prep_weights<<<(prep_threads + 255) / 256, 256, 0, stream>>>(w1, w2, W1s, W2s);
    nf_fused<<<n / TILE_M, THREADS, 0, stream>>>(coords, table, b1, b2,
                                                 W1s, W2s, out, lp);
}
